// RNN_76347338654361
// MI455X (gfx1250) — matrix operand order repaired and checked by reading, not compiled, not run on hardware
//
#include <hip/hip_runtime.h>

// ---------------------------------------------------------------------------
// 3-layer GRU language model, persistent-kernel recurrence for MI455X (gfx1250)
//
// Roofline: sequential over T=1024; per step ~1.2 GFLOP of [128x1536x512]
// GEMMs. Weights (19MB) are L2-resident; HBM traffic is trivial -> the knobs
// that matter are matrix-core precision (bf16 WMMA, f32 accum), sync depth
// (3 grid barriers per step), and LDS reuse of weight tiles (8 waves share).
// Weight tiles stream global->LDS with GLOBAL_LOAD_ASYNC_TO_LDS_B128
// (ASYNCcnt) into a double-buffered LDS tile, one K-chunk ahead of the WMMAs.
// ---------------------------------------------------------------------------

#define B_    128
#define T_    1024
#define V_    65
#define VPAD_ 80
#define H_    512
#define L_    3
#define G3H_  1536
#define NWG_  32       // 32 WGs x 16 cols = H; all co-resident -> barrier safe
#define SROW_ 40       // LDS row stride (elements): 80B, spreads banks
#define SBUF_ (48 * SROW_)

typedef __attribute__((ext_vector_type(16))) __bf16 v16bf;
typedef __attribute__((ext_vector_type(8)))  __bf16 v8bf;
typedef __attribute__((ext_vector_type(8)))  float  v8f;
typedef int v4i_vs __attribute__((vector_size(4 * sizeof(int))));

#if defined(__gfx1250__) && __has_builtin(__builtin_amdgcn_global_load_async_to_lds_b128)
#define HAVE_ASYNC_LDS 1
#else
#define HAVE_ASYNC_LDS 0
#endif

__device__ __forceinline__ v8f zero8() {
  v8f z = {0.f, 0.f, 0.f, 0.f, 0.f, 0.f, 0.f, 0.f};
  return z;
}

__device__ __forceinline__ v16bf cat16(v8bf lo, v8bf hi) {
  return __builtin_shufflevector(lo, hi, 0, 1, 2, 3, 4, 5, 6, 7,
                                         8, 9, 10, 11, 12, 13, 14, 15);
}

__device__ __forceinline__ v8bf ld8(const __bf16* p) {
  return *(const v8bf*)p;            // 16B -> global_load_b128 / ds_load_b128
}

__device__ __forceinline__ v8bf cvt8(const float* p) {
  float4 a = *(const float4*)p;
  float4 b = *(const float4*)(p + 4);
  v8bf r;
  r[0] = (__bf16)a.x; r[1] = (__bf16)a.y; r[2] = (__bf16)a.z; r[3] = (__bf16)a.w;
  r[4] = (__bf16)b.x; r[5] = (__bf16)b.y; r[6] = (__bf16)b.z; r[7] = (__bf16)b.w;
  return r;
}

__device__ __forceinline__ float sigm_f(float x) {
  return 1.0f / (1.0f + __expf(-x));
}
__device__ __forceinline__ float tanh_f(float x) {
  x = fminf(fmaxf(x, -12.0f), 12.0f);
  float e = __expf(-2.0f * x);
  return (1.0f - e) / (1.0f + e);
}

__device__ __forceinline__ v8f wmma_bf(v16bf a, v16bf b, v8f c) {
  // (neg_a, A, neg_b, B, c_mod, C, reuse_a, reuse_b)
  return __builtin_amdgcn_wmma_f32_16x16x32_bf16(false, a, false, b,
                                                 (short)0, c, false, false);
}

#if HAVE_ASYNC_LDS
__device__ __forceinline__ void async_cp16(const __bf16* g, __bf16* l) {
  // global (AS1) -> LDS (AS3), 16B per active lane, tracked by ASYNCcnt.
  // Builtin signature (from clang diagnostic): (v4i AS1*, v4i AS3*, imm, imm).
  __attribute__((address_space(3))) __bf16* l3 =
      (__attribute__((address_space(3))) __bf16*)l;           // addrspacecast
  __builtin_amdgcn_global_load_async_to_lds_b128(
      (__attribute__((address_space(1))) v4i_vs*)(uintptr_t)g,
      (__attribute__((address_space(3))) v4i_vs*)l3,
      0, 0);
}
__device__ __forceinline__ void wait_async0() {
#if __has_builtin(__builtin_amdgcn_s_wait_asynccnt)
  __builtin_amdgcn_s_wait_asynccnt(0);
#else
  asm volatile("s_wait_asynccnt 0x0" ::: "memory");
#endif
}
#endif

// Grid-wide barrier: monotonically increasing generation count, counter reset
// by prep_kernel every launch -> no cross-call state.
__device__ __forceinline__ void grid_sync(unsigned* bar, unsigned gen) {
  __syncthreads();
  if (threadIdx.x == 0) {
    __threadfence();
    __hip_atomic_fetch_add(bar, 1u, __ATOMIC_RELEASE, __HIP_MEMORY_SCOPE_AGENT);
    const unsigned target = gen * NWG_;
    while (__hip_atomic_load(bar, __ATOMIC_ACQUIRE, __HIP_MEMORY_SCOPE_AGENT) < target) {
      __builtin_amdgcn_s_sleep(2);
    }
    __threadfence();
  }
  __syncthreads();
}

// ---------------------------------------------------------------------------
// Prep: convert weights to bf16 (W_out padded to 80 rows), zero h-state
// buffer #1 (read at t=0), reset barrier counter. Re-done every launch.
// ---------------------------------------------------------------------------
__global__ void prep_kernel(const float* __restrict__ w_ih,
                            const float* __restrict__ w_hh,
                            const float* __restrict__ w_out,
                            __bf16* __restrict__ wihb,
                            __bf16* __restrict__ whhb,
                            __bf16* __restrict__ woutb,
                            float*  __restrict__ hF1,
                            __bf16* __restrict__ hB1,
                            unsigned* __restrict__ bar) {
  size_t i0 = (size_t)blockIdx.x * blockDim.x + threadIdx.x;
  size_t st = (size_t)gridDim.x * blockDim.x;
  const size_t NW = (size_t)L_ * G3H_ * H_;
  for (size_t i = i0; i < NW; i += st) {
    wihb[i] = (__bf16)w_ih[i];
    whhb[i] = (__bf16)w_hh[i];
  }
  for (size_t i = i0; i < (size_t)VPAD_ * H_; i += st) {
    size_t r = i / H_;
    woutb[i] = (r < V_) ? (__bf16)w_out[i] : (__bf16)0.0f;
  }
  for (size_t i = i0; i < (size_t)L_ * B_ * H_; i += st) {
    hF1[i] = 0.0f;
    hB1[i] = (__bf16)0.0f;
  }
  if (i0 == 0) *bar = 0u;
}

// ---------------------------------------------------------------------------
// Persistent GRU kernel. Grid = 32 WGs x 256 threads (8 wave32).
//   WG g owns h columns [g*16, g*16+16); wave w owns batch rows [w*16, w*16+16).
//   Per layer-step each wave does 96 v_wmma_f32_16x16x32_bf16.
// Double-buffered h (f32 for update, bf16 for matmul), parity = t & 1.
// Weight tiles (48 W_ih + 48 W_hh rows x 32 K) double-buffered in LDS and
// streamed with async global->LDS copies one K-chunk ahead.
// ---------------------------------------------------------------------------
__global__ __launch_bounds__(256)
void gru_persistent(const int*    __restrict__ x_seq,
                    const float*  __restrict__ emb,
                    const float*  __restrict__ bias,   // [L][3H]
                    const float*  __restrict__ bn,     // [L][H]
                    const float*  __restrict__ b_out,  // [V]
                    const __bf16* __restrict__ wihb,   // [L][3H][H]
                    const __bf16* __restrict__ whhb,   // [L][3H][H]
                    const __bf16* __restrict__ woutb,  // [VPAD][H]
                    float*        __restrict__ hf32,   // [2][L][B][H]
                    __bf16*       __restrict__ hbf,    // [2][L][B][H]
                    float*        __restrict__ out,    // [B][T][V]
                    unsigned*     __restrict__ bar) {
  __shared__ __bf16 sIh[2][SBUF_];
  __shared__ __bf16 sHh[2][SBUF_];

  const int wg   = blockIdx.x;
  const int tid  = threadIdx.x;
  const int lane = tid & 31;
  const int wv   = tid >> 5;        // wave 0..7 -> M tile
  const int l16  = lane & 15;
  const int half = lane >> 4;       // K-half selector (this code's contiguous K order)
  const int mrow = wv * 16 + l16;   // batch row this lane fetches A data for

  // Stager mapping: threads 0..191 each own one contiguous 32B chunk of the
  // 96-row x 64B weight tile (row = q>>1, 32B half = q&1).
  const bool stager = (tid < 192);
  const int  srow   = tid >> 1;                     // 0..95 (clamped by stager)
  const int  sh32   = tid & 1;
  const int  rr     = (srow < 48) ? srow : srow - 48;
  const int  wcol   = (rr >> 4) * H_ + wg * 16 + (rr & 15);  // W row (gate blk)
  __bf16* sdst = ((srow < 48) ? &sIh[0][0] : &sHh[0][0]) + rr * SROW_ + sh32 * 16;

  unsigned gen = 0;

  for (int t = 0; t < T_; ++t) {
    const int cur = t & 1;
    const int old = cur ^ 1;

    for (int l = 0; l < L_; ++l) {
      const __bf16* Wih  = wihb + (size_t)l * G3H_ * H_;
      const __bf16* Whh  = whhb + (size_t)l * G3H_ * H_;
      const float*  hOldF = hf32 + ((size_t)old * L_ + l) * B_ * H_;
      const __bf16* hOldB = hbf  + ((size_t)old * L_ + l) * B_ * H_;
      float*        hNewF = hf32 + ((size_t)cur * L_ + l) * B_ * H_;
      __bf16*       hNewB = hbf  + ((size_t)cur * L_ + l) * B_ * H_;

      // Per-layer source pointer for this thread's staged chunk.
      const __bf16* srcBase =
          ((srow < 48) ? Wih : Whh) + (size_t)wcol * H_ + sh32 * 16;

      // Layer input: l==0 gathers embedding rows on the fly (f32 -> bf16),
      // l>0 reads the previous layer's fresh bf16 hidden state.
      const float*  embRow = nullptr;
      const __bf16* xRow   = nullptr;
      if (l == 0) {
        int s  = x_seq[mrow * T_ + t];
        embRow = emb + (size_t)s * H_;
      } else {
        xRow = hbf + ((size_t)cur * L_ + (l - 1)) * B_ * H_ + (size_t)mrow * H_;
      }
      const __bf16* hRow = hOldB + (size_t)mrow * H_;

      v8bf stg0, stg1;  // fallback register staging
      auto stage_issue = [&](int k0, int buf) {
        if (stager) {
          const __bf16* s = srcBase + k0;
#if HAVE_ASYNC_LDS
          __bf16* d = sdst + buf * SBUF_;
          async_cp16(s, d);
          async_cp16(s + 8, d + 8);
#else
          stg0 = ld8(s);
          stg1 = ld8(s + 8);
          (void)buf;
#endif
        }
      };
      auto stage_commit = [&](int buf) {
#if HAVE_ASYNC_LDS
        wait_async0();
        (void)buf;
#else
        if (stager) {
          __bf16* d = sdst + buf * SBUF_;
          *(v8bf*)d = stg0;
          *(v8bf*)(d + 8) = stg1;
        }
#endif
      };

      v8f aR = zero8(), aZ = zero8(), aIN = zero8(), aHN = zero8();

      // Prologue: stage K-chunk 0 into LDS buffer 0.
      stage_issue(0, 0);
      stage_commit(0);
      __syncthreads();

      for (int k0 = 0, it = 0; k0 < H_; k0 += 32, ++it) {
        const int  bufc = it & 1;
        const bool more = (k0 + 32) < H_;
        // Kick next chunk's async copies before doing this chunk's math.
        if (more) stage_issue(k0 + 32, bufc ^ 1);

        // A operands (contiguous K order, matching B): lane(m,half) holds
        // K = k0+16*half+[0..15].
        const int ka = k0 + half * 16;
        v16bf Ax, Ah;
        if (l == 0) {
          Ax = cat16(cvt8(embRow + ka), cvt8(embRow + ka + 8));
        } else {
          Ax = cat16(ld8(xRow + ka), ld8(xRow + ka + 8));
        }
        Ah = cat16(ld8(hRow + ka), ld8(hRow + ka + 8));

        // B operands (32x16): lane = out column, half picks K 0-15/16-31,
        // 16 contiguous bf16 of the (transposed-use) weight row.
        const __bf16* bIh = &sIh[bufc][0];
        const __bf16* bHh = &sHh[bufc][0];
        const int boff = half * 16;
        auto ldB = [&](const __bf16* s, int rloc) -> v16bf {
          const __bf16* p = s + rloc * SROW_ + boff;
          return cat16(*(const v8bf*)p, *(const v8bf*)(p + 8));
        };
        // r gate: i and h parts summed in one accumulator
        aR = wmma_bf(Ax, ldB(bIh, 0 * 16 + l16), aR);
        aR = wmma_bf(Ah, ldB(bHh, 0 * 16 + l16), aR);
        // z gate
        aZ = wmma_bf(Ax, ldB(bIh, 1 * 16 + l16), aZ);
        aZ = wmma_bf(Ah, ldB(bHh, 1 * 16 + l16), aZ);
        // n gate: i_n and h_n must stay separate (n = tanh(i_n + r*(h_n+b_n)))
        aIN = wmma_bf(Ax, ldB(bIh, 2 * 16 + l16), aIN);
        aHN = wmma_bf(Ah, ldB(bHh, 2 * 16 + l16), aHN);

        // Land the prefetched chunk, then make it visible to all 8 waves.
        if (more) stage_commit(bufc ^ 1);
        __syncthreads();
      }

      // Elementwise GRU update: C-layout gives this lane 8 rows of one column.
      const int   col = wg * 16 + l16;
      const float brr = bias[l * G3H_ + col];
      const float bzz = bias[l * G3H_ + H_ + col];
      const float bin = bias[l * G3H_ + 2 * H_ + col];
      const float bnn = bn[l * H_ + col];
#pragma unroll
      for (int j = 0; j < 8; ++j) {
        int   row = wv * 16 + j + half * 8;
        float r   = sigm_f(aR[j] + brr);
        float z   = sigm_f(aZ[j] + bzz);
        float n   = tanh_f(aIN[j] + bin + r * (aHN[j] + bnn));
        float hp  = hOldF[(size_t)row * H_ + col];
        float hn  = n + z * (hp - n);
        hNewF[(size_t)row * H_ + col] = hn;
        hNewB[(size_t)row * H_ + col] = (__bf16)hn;
      }
      grid_sync(bar, ++gen);
    }

    // Output projection y(t) = h2(t) @ W_out^T + b_out on WGs 0..4 (80 cols,
    // 65 valid). No trailing barrier needed: the next conflicting write to
    // hbf[cur][2] is ordered behind a barrier these WGs must pass first.
    if (wg < 5) {
      const __bf16* hRow = hbf + ((size_t)cur * L_ + 2) * B_ * H_ + (size_t)mrow * H_;
      const int vcol = wg * 16 + l16;
      v8f acc = zero8();
      for (int k0 = 0; k0 < H_; k0 += 32) {
        const int ka = k0 + half * 16;
        v16bf Ah = cat16(ld8(hRow + ka), ld8(hRow + ka + 8));
        const __bf16* wr = woutb + (size_t)vcol * H_ + k0 + half * 16;
        v16bf Bv = cat16(ld8(wr), ld8(wr + 8));
        acc = wmma_bf(Ah, Bv, acc);
      }
      if (vcol < V_) {
        float bo = b_out[vcol];
#pragma unroll
        for (int j = 0; j < 8; ++j) {
          int row = wv * 16 + j + half * 8;
          out[((size_t)row * T_ + t) * V_ + vcol] = acc[j] + bo;
        }
      }
    }
  }
}

// ---------------------------------------------------------------------------
extern "C" void kernel_launch(void* const* d_in, const int* in_sizes, int n_in,
                              void* d_out, int out_size, void* d_ws, size_t ws_size,
                              hipStream_t stream) {
  (void)in_sizes; (void)n_in; (void)out_size; (void)ws_size;
  const int*   x_seq = (const int*)  d_in[0];
  const float* emb   = (const float*)d_in[1];
  const float* w_ih  = (const float*)d_in[2];
  const float* w_hh  = (const float*)d_in[3];
  const float* b     = (const float*)d_in[4];
  const float* b_n   = (const float*)d_in[5];
  const float* w_out = (const float*)d_in[6];
  const float* b_out = (const float*)d_in[7];
  float* out = (float*)d_out;

  // Workspace layout (256B aligned), ~11.9 MB total.
  char*  ws  = (char*)d_ws;
  size_t off = 0;
  auto take = [&](size_t bytes) -> char* {
    char* p = ws + off;
    off = (off + bytes + 255) & ~(size_t)255;
    return p;
  };
  unsigned* bar  = (unsigned*)take(256);
  float*    hf32 = (float*)  take((size_t)2 * L_ * B_ * H_ * sizeof(float));
  __bf16*   hbf  = (__bf16*) take((size_t)2 * L_ * B_ * H_ * sizeof(__bf16));
  __bf16*   wihb = (__bf16*) take((size_t)L_ * G3H_ * H_ * sizeof(__bf16));
  __bf16*   whhb = (__bf16*) take((size_t)L_ * G3H_ * H_ * sizeof(__bf16));
  __bf16*   woutb= (__bf16*) take((size_t)VPAD_ * H_ * sizeof(__bf16));

  prep_kernel<<<2048, 256, 0, stream>>>(
      w_ih, w_hh, w_out, wihb, whhb, woutb,
      hf32 + (size_t)L_ * B_ * H_,   // zero the t=0 "old" buffer (index 1)
      hbf  + (size_t)L_ * B_ * H_,
      bar);

  gru_persistent<<<NWG_, 256, 0, stream>>>(
      x_seq, emb, b, b_n, b_out, wihb, whhb, woutb, hf32, hbf, out, bar);
}
